// ViewGNN_34291018891309
// MI455X (gfx1250) — matrix operand order repaired and checked by reading, not compiled, not run on hardware
//
#include <hip/hip_runtime.h>
#include <hip/hip_bf16.h>
#include <math.h>

// =====================================================================
// ViewGNN forward for MI455X (gfx1250, wave32, WMMA).
// All dense GEMMs (N=32 outputs) use v_wmma_f32_16x16x32_bf16 with fp32
// accumulation; fp32 inputs are converted to bf16 on the fly into LDS
// (GEMM is HBM-bound: 128MB of x @ 23.3TB/s dwarfs the 2.1 GFLOP).
// WMMA fragments are fetched as ds_load_b128 pairs: A's per-lane layout
// is two contiguous 8-elem bf16 runs; B is staged pre-swizzled into
// fragment order so each lane's 16 values are contiguous.
// Edge aggregation uses float4 gathers + global_atomic_add_f32 (agg
// arrays fit in the 192MB L2, so the RMW stays on-chip).
// Top-k is a deterministic bitonic sort (n is a power of two).
// Requires ws_size >= ~105 MB.
// =====================================================================

typedef __attribute__((ext_vector_type(16))) __bf16 v16bf;
typedef __attribute__((ext_vector_type(8)))  __bf16 v8bf;
typedef __attribute__((ext_vector_type(8)))  float  v8f;

__device__ __forceinline__ unsigned short f2bf_bits(float f) {
  union { float f; unsigned int u; } v; v.f = f;
  unsigned int u = v.u;
  u += 0x7FFFu + ((u >> 16) & 1u);   // round-to-nearest-even
  return (unsigned short)(u >> 16);
}
__device__ __forceinline__ unsigned int pack2bf(float a, float b) {
  return (unsigned int)f2bf_bits(a) | ((unsigned int)f2bf_bits(b) << 16);
}
__device__ __forceinline__ float lrelu(float v) { return v > 0.f ? v : 0.2f * v; }

// ---------------------------------------------------------------------
// GEMM: C[M,32] = A[M,K](fp32) * W[K,32](fp32 -> bf16 in LDS)
// 256 threads = 8 waves; 4 M-tiles x 2 N-tiles => 64x32 per block.
// sA: row-major [64][32] bf16. sB: fragment-swizzled
//   sB[tile*1024 + (hl*32+col)*16 + j] = B[tile*32 + hl*16 + j][col]
// so every lane's A and B fragments are two 16B LDS vectors each.
// ---------------------------------------------------------------------
__global__ __launch_bounds__(256)
void gemm32_wmma(const float* __restrict__ A, const float* __restrict__ Bw,
                 const float* __restrict__ bias, float* __restrict__ C,
                 int M, int K) {
  __shared__ alignas(16) unsigned short sB[512 * 32];   // K<=512
  __shared__ alignas(16) unsigned short sA[64 * 32];
  const int tid  = threadIdx.x;
  const int wave = tid >> 5, lane = tid & 31;
  const int mw = wave >> 1, nw = wave & 1;
  const int hl = lane >> 4, r = lane & 15;

  // Stage B panel, converted + swizzled into fragment order (packed stores).
  for (int i = tid; i < K * 16; i += 256) {             // exact multiple of 256
    int tile = i >> 9;                                  // 512 dword-pairs per k-tile
    int rem  = i & 511;
    int grp  = rem >> 3;                                // hl*32 + col  (0..63)
    int jp   = rem & 7;                                 // j pair
    int hh   = grp >> 5, col = grp & 31;
    int k    = tile * 32 + hh * 16 + jp * 2;
    ((unsigned int*)sB)[i] = pack2bf(Bw[k * 32 + col], Bw[(k + 1) * 32 + col]);
  }

  v8f acc = {};
  const size_t rowBase = (size_t)blockIdx.x * 64;
  const bool fullTile = (rowBase + 64 <= (size_t)M);    // uniform across block

  for (int k0 = 0; k0 < K; k0 += 32) {
    if (fullTile) {
      // fast path: vector loads, packed bf16 stores; exact 4 iterations
      for (int i = tid; i < 64 * 16; i += 256) {
        int rr = i >> 4, cp = i & 15;
        float2 v = *(const float2*)&A[(rowBase + rr) * K + k0 + cp * 2];
        ((unsigned int*)sA)[i] = pack2bf(v.x, v.y);
      }
    } else {
      for (int i = tid; i < 64 * 32; i += 256) {        // exact 8 iterations
        int rr = i >> 5, cc = i & 31;
        size_t grow = rowBase + (size_t)rr;
        float v = (grow < (size_t)M) ? A[grow * K + k0 + cc] : 0.f;
        sA[i] = f2bf_bits(v);
      }
    }
    if (k0 + 32 < K) {                                  // prefetch next A k-tile
      size_t grow = rowBase + (size_t)(tid >> 2);
      if (grow < (size_t)M)
        __builtin_prefetch(A + grow * K + k0 + 32 + (size_t)(tid & 3) * 8, 0, 1);
    }
    __syncthreads();

    // A fragment: contiguous 16 elems, k = 16*hl + s (ds_load_b128 x2)
    const int abase = (mw * 16 + r) * 32 + hl * 16;
    v8bf alo = *(const v8bf*)&sA[abase];
    v8bf ahi = *(const v8bf*)&sA[abase + 8];
    v16bf afrag = __builtin_shufflevector(alo, ahi,
        0,1,2,3,4,5,6,7,8,9,10,11,12,13,14,15);

    // B fragment: contiguous 16 elems in swizzled layout (ds_load_b128 x2)
    const int bbase = (k0 >> 5) * 1024 + (hl * 32 + nw * 16 + r) * 16;
    v8bf blo = *(const v8bf*)&sB[bbase];
    v8bf bhi = *(const v8bf*)&sB[bbase + 8];
    v16bf bfrag = __builtin_shufflevector(blo, bhi,
        0,1,2,3,4,5,6,7,8,9,10,11,12,13,14,15);

    acc = __builtin_amdgcn_wmma_f32_16x16x32_bf16(
        false, afrag, false, bfrag, (short)0, acc, false, false);
    __syncthreads();
  }

#pragma unroll
  for (int i = 0; i < 8; ++i) {                         // C: VGPR i -> M = i + 8*hl
    size_t grow = rowBase + (size_t)(mw * 16 + hl * 8 + i);
    int col = nw * 16 + r;
    if (grow < (size_t)M) {
      float bb = bias ? bias[col] : 0.f;
      C[grow * 32 + col] = acc[i] + bb;
    }
  }
}

// ------------------------- graph helpers ----------------------------
__global__ void fill_f(float* p, float v, int n) {
  int t = blockIdx.x * blockDim.x + threadIdx.x; if (t < n) p[t] = v;
}
__global__ void fill_i(int* p, int v, int n) {
  int t = blockIdx.x * blockDim.x + threadIdx.x; if (t < n) p[t] = v;
}
__global__ void iota_i(int* p, int n) {
  int t = blockIdx.x * blockDim.x + threadIdx.x; if (t < n) p[t] = t;
}
__global__ void copy_f(const float* a, float* b, int n) {
  int t = blockIdx.x * blockDim.x + threadIdx.x; if (t < n) b[t] = a[t];
}

__global__ void degree_kernel(const int* __restrict__ dst, const float* __restrict__ w,
                              float* __restrict__ deg, int nE) {
  int e = blockIdx.x * blockDim.x + threadIdx.x;
  if (e < nE) atomicAdd(&deg[dst[e]], w ? w[e] : 1.0f);
}
__global__ void deg_to_dinv(float* deg, int n) {   // in-place: dinv = rsqrt(deg+1)
  int t = blockIdx.x * blockDim.x + threadIdx.x;
  if (t < n) deg[t] = rsqrtf(deg[t] + 1.0f);
}

// agg[dst] += norm * h[src];  norm = useNorm ? dinv[s]*dinv[d]*w : w
__global__ void edge_aggregate(const int* __restrict__ src, const int* __restrict__ dst,
                               const float* __restrict__ w, const float* __restrict__ dinv,
                               const float* __restrict__ h, float* __restrict__ agg,
                               int nE, int useNorm) {
  long long t = (long long)blockIdx.x * blockDim.x + threadIdx.x;
  int e = (int)(t >> 3);
  if (e >= nE) return;
  int g = (int)(t & 7);
  float we = w ? w[e] : 1.0f;
  if (we == 0.f) return;
  int s = src[e], d = dst[e];
  float norm = useNorm ? dinv[s] * dinv[d] * we : we;
  float4 v = ((const float4*)h)[(size_t)s * 8 + g];
  float* ap = agg + (size_t)d * 32 + g * 4;
  atomicAdd(ap + 0, norm * v.x); atomicAdd(ap + 1, norm * v.y);
  atomicAdd(ap + 2, norm * v.z); atomicAdd(ap + 3, norm * v.w);
}

// out = lrelu(agg + dinv^2 * h + bias)
__global__ void gcn_finalize(const float* __restrict__ agg, const float* __restrict__ h,
                             const float* __restrict__ dinv, const float* __restrict__ bias,
                             float* __restrict__ out, int n) {
  long long t = (long long)blockIdx.x * blockDim.x + threadIdx.x;
  if (t < (long long)n * 32) {
    int i = (int)(t >> 5), f = (int)(t & 31);
    float di = dinv[i];
    out[t] = lrelu(agg[t] + di * di * h[t] + bias[f]);
  }
}

__global__ void batch_rowmax(const float* __restrict__ x, float* __restrict__ out, int npb) {
  __shared__ float red[256];
  int f = threadIdx.x & 31, c = threadIdx.x >> 5;
  float m = -INFINITY;
  for (int r = c; r < npb; r += 8)
    m = fmaxf(m, x[((size_t)blockIdx.x * npb + r) * 32 + f]);
  red[threadIdx.x] = m; __syncthreads();
  if (c == 0) {
    for (int i = 1; i < 8; ++i) m = fmaxf(m, red[i * 32 + f]);
    out[blockIdx.x * 32 + f] = m;
  }
}
__global__ void batch_rowmean(const float* __restrict__ x, float* __restrict__ out, int npb) {
  __shared__ float red[256];
  int f = threadIdx.x & 31, c = threadIdx.x >> 5;
  float s = 0.f;
  for (int r = c; r < npb; r += 8)
    s += x[((size_t)blockIdx.x * npb + r) * 32 + f];
  red[threadIdx.x] = s; __syncthreads();
  if (c == 0) {
    for (int i = 1; i < 8; ++i) s += red[i * 32 + f];
    out[blockIdx.x * 32 + f] = s / (float)npb;
  }
}

// ------------------------- virtual node -----------------------------
__global__ void gen_vnode_edges(int* src, int* dst, int n_per, int bsz) {
  int e = blockIdx.x * blockDim.x + threadIdx.x;
  if (e < n_per * bsz) { src[e] = e / bsz; dst[e] = n_per; }  // jnp.repeat semantics
}
__global__ void build_vnode_x(const float* __restrict__ x, float* __restrict__ xv,
                              int n_per, int bsz) {
  long long t = (long long)blockIdx.x * blockDim.x + threadIdx.x;
  long long nv = (long long)(n_per + 1) * bsz;
  if (t < nv * 32) {
    int row = (int)(t >> 5), f = (int)(t & 31);
    int b = row / (n_per + 1), r = row % (n_per + 1);
    xv[t] = (r < n_per) ? x[((size_t)b * n_per + r) * 32 + f] : 0.f;
  }
}
__global__ void extract_last(const float* __restrict__ xg, float* __restrict__ out,
                             int n_per, int bsz) {
  int t = blockIdx.x * blockDim.x + threadIdx.x;
  if (t < bsz * 32) {
    int b = t >> 5, f = t & 31;
    out[t] = xg[((size_t)b * (n_per + 1) + n_per) * 32 + f];
  }
}

// ------------------------- SAG pooling ------------------------------
__global__ void score_kernel(const float* __restrict__ sagg, const float* __restrict__ x,
                             const float* __restrict__ wrel, const float* __restrict__ brel,
                             const float* __restrict__ wroot, float* __restrict__ score, int n) {
  int i = blockIdx.x * blockDim.x + threadIdx.x;
  if (i < n) {
    float s = brel[0];
#pragma unroll
    for (int f = 0; f < 32; ++f)
      s += sagg[(size_t)i * 32 + f] * wrel[f] + x[(size_t)i * 32 + f] * wroot[f];
    score[i] = s;
  }
}

// descending order, ties by ascending index (matches lax.top_k)
__global__ void bitonic_step(float* __restrict__ key, int* __restrict__ idx,
                             int n, int k, int j) {
  int i = blockIdx.x * blockDim.x + threadIdx.x;
  int ixj = i ^ j;
  if (i < n && ixj > i && ixj < n) {
    float a = key[i], b = key[ixj];
    int ia = idx[i], ib = idx[ixj];
    bool before = (a > b) || (a == b && ia < ib);
    bool dir = ((i & k) == 0);
    if (dir ? !before : before) {
      key[i] = b; key[ixj] = a; idx[i] = ib; idx[ixj] = ia;
    }
  }
}

__global__ void pool_gather(const float* __restrict__ x, const float* __restrict__ skey,
                            const int* __restrict__ sidx, float* __restrict__ xo, int k) {
  long long t = (long long)blockIdx.x * blockDim.x + threadIdx.x;
  if (t < (long long)k * 32) {
    int r = (int)(t >> 5), f = (int)(t & 31);
    xo[t] = x[(size_t)sidx[r] * 32 + f] * tanhf(skey[r]);
  }
}
__global__ void scatter_inv(const int* __restrict__ sidx, int* __restrict__ inv, int k) {
  int r = blockIdx.x * blockDim.x + threadIdx.x;
  if (r < k) inv[sidx[r]] = r;
}
__global__ void edge_remap(const int* __restrict__ src, const int* __restrict__ dst,
                           const float* __restrict__ w, const int* __restrict__ inv,
                           int* __restrict__ s2, int* __restrict__ d2,
                           float* __restrict__ w2, int nE) {
  int e = blockIdx.x * blockDim.x + threadIdx.x;
  if (e < nE) {
    int ns = inv[src[e]], nd = inv[dst[e]];
    float we = w ? w[e] : 1.0f;
    bool valid = (ns >= 0) && (nd >= 0) && (we > 0.f);
    s2[e] = valid ? ns : 0; d2[e] = valid ? nd : 0; w2[e] = valid ? 1.f : 0.f;
  }
}

// ------------------------- heads ------------------------------------
__global__ void small_linear(const float* __restrict__ X, const float* __restrict__ W,
                             const float* __restrict__ b, float* __restrict__ Y,
                             int M, int K, int N, int act) {
  int t = blockIdx.x * blockDim.x + threadIdx.x;
  if (t < M * N) {
    int m = t / N, n = t % N;
    float s = b ? b[n] : 0.f;
    for (int k = 0; k < K; ++k) s += X[m * K + k] * W[k * N + n];
    Y[t] = act ? lrelu(s) : s;
  }
}
__global__ void concat2(const float* A, const float* Bp, float* Y, int M, int K1, int K2) {
  int t = blockIdx.x * blockDim.x + threadIdx.x;
  if (t < M * (K1 + K2)) {
    int m = t / (K1 + K2), j = t % (K1 + K2);
    Y[t] = (j < K1) ? A[m * K1 + j] : Bp[m * K2 + (j - K1)];
  }
}
__global__ void pack_out(const float* o2, const float* xl, const float* xg, float* o) {
  int t = blockIdx.x * blockDim.x + threadIdx.x;
  if (t < 106) {
    if (t < 32) o[t] = o2[t];
    else if (t < 64) o[t] = xl[t - 32];
    else if (t < 96) o[t] = xg[t - 64];
    else o[t] = 1.0f;
  }
}

// =====================================================================
extern "C" void kernel_launch(void* const* d_in, const int* in_sizes, int n_in,
                              void* d_out, int out_size, void* d_ws, size_t ws_size,
                              hipStream_t stream) {
  (void)n_in; (void)out_size; (void)ws_size;
  const int Bsz = 16;
  const int NN  = in_sizes[2];          // 65536
  const int D   = in_sizes[0] / NN;     // 512
  const int E   = in_sizes[1] / 2;      // 2097152
  const int Npb = NN / Bsz;             // 4096

  const float* x    = (const float*)d_in[0];
  const int*   ei   = (const int*)d_in[1];
  const int*   srcp = ei;
  const int*   dstp = ei + E;
  const float* W_lc1 = (const float*)d_in[3];  const float* b_lc1 = (const float*)d_in[4];
  const float* W_lc2 = (const float*)d_in[5];  const float* b_lc2 = (const float*)d_in[6];
  const float* W_gc1 = (const float*)d_in[7];  const float* b_gc1 = (const float*)d_in[8];
  const float* W_gc2 = (const float*)d_in[9];  const float* b_gc2 = (const float*)d_in[10];
  const float* Wrel1 = (const float*)d_in[11]; const float* brel1 = (const float*)d_in[12];
  const float* Wroot1= (const float*)d_in[13];
  const float* Wrel2 = (const float*)d_in[14]; const float* brel2 = (const float*)d_in[15];
  const float* Wroot2= (const float*)d_in[16];
  const float* W_l1  = (const float*)d_in[17]; const float* b_l1 = (const float*)d_in[18];
  const float* W_l2  = (const float*)d_in[19]; const float* b_l2 = (const float*)d_in[20];
  const float* W_ll1 = (const float*)d_in[21]; const float* b_ll1 = (const float*)d_in[22];
  const float* W_ll2 = (const float*)d_in[23]; const float* b_ll2 = (const float*)d_in[24];
  const float* W_gl1 = (const float*)d_in[25]; const float* b_gl1 = (const float*)d_in[26];
  const float* W_gl2 = (const float*)d_in[27]; const float* b_gl2 = (const float*)d_in[28];

  // ---- workspace bump allocator ----
  char* base = (char*)d_ws; size_t off = 0;
  auto alloc = [&](size_t bytes) -> void* {
    void* p = base + off; off += (bytes + 255) & ~(size_t)255; return p;
  };
  const size_t NNH = (size_t)NN * 32;
  float* h1   = (float*)alloc(NNH * 4);
  float* agg1 = (float*)alloc(NNH * 4);
  float* x1   = (float*)alloc(NNH * 4);
  float* deg  = (float*)alloc((size_t)NN * 4);
  float* score= (float*)alloc((size_t)NN * 4);
  float* skey = (float*)alloc((size_t)NN * 4);
  int*   sidx = (int*)alloc((size_t)NN * 4);
  int*   inv  = (int*)alloc((size_t)NN * 4);
  int*   s2   = (int*)alloc((size_t)E * 4);
  int*   d2   = (int*)alloc((size_t)E * 4);
  float* w2   = (float*)alloc((size_t)E * 4);
  const int k1 = NN / 2, k2n = NN / 4;
  float* x2   = (float*)alloc((size_t)k1 * 32 * 4);
  float* h2   = (float*)alloc((size_t)k1 * 32 * 4);
  float* agg2 = (float*)alloc((size_t)k1 * 32 * 4);
  float* x2p  = (float*)alloc((size_t)k1 * 32 * 4);
  float* x3   = (float*)alloc((size_t)k2n * 32 * 4);
  const int nv1 = (Npb + 1) * Bsz, Ev1 = Npb * Bsz;
  float* xv   = (float*)alloc((size_t)nv1 * 32 * 4);
  float* hv   = (float*)alloc((size_t)nv1 * 32 * 4);
  float* aggv = (float*)alloc((size_t)nv1 * 32 * 4);
  float* xgv  = (float*)alloc((size_t)nv1 * 32 * 4);
  float* degv = (float*)alloc((size_t)nv1 * 4);
  int*   vsrc = (int*)alloc((size_t)Ev1 * 4);
  int*   vdst = (int*)alloc((size_t)Ev1 * 4);
  float* xl1  = (float*)alloc(Bsz * 32 * 4);
  float* xl2  = (float*)alloc(Bsz * 32 * 4);
  float* xg1  = (float*)alloc(Bsz * 32 * 4);
  float* xg2  = (float*)alloc(Bsz * 32 * 4);
  float* xm   = (float*)alloc(Bsz * 32 * 4);
  float* cat64a = (float*)alloc(16 * 64 * 4);
  float* cat64b = (float*)alloc(16 * 64 * 4);
  float* t8   = (float*)alloc(16 * 8 * 4);
  float* obuf = (float*)alloc(16 * 2 * 4);
  float* xlbuf= (float*)alloc(16 * 2 * 4);
  float* xgbuf= (float*)alloc(16 * 2 * 4);

  auto cdiv = [](long long a, long long b) { return (int)((a + b - 1) / b); };
  auto zero_f = [&](float* p, long long n) { fill_f<<<cdiv(n,256),256,0,stream>>>(p, 0.f, (int)n); };
  auto bitonic = [&](float* key, int* idx, int n) {
    int blocks = cdiv(n, 256);
    for (int k = 2; k <= n; k <<= 1)
      for (int j = k >> 1; j > 0; j >>= 1)
        bitonic_step<<<blocks, 256, 0, stream>>>(key, idx, n, k, j);
  };
  // full GCN conv: out = lrelu(agg + dinv^2 h + b)
  auto gcn = [&](const float* xin, const int* s, const int* d, const float* w,
                 const float* Wm, const float* bm, int n, int K, int nE,
                 float* hbuf, float* abuf, float* dg, float* out) {
    gemm32_wmma<<<cdiv(n,64),256,0,stream>>>(xin, Wm, nullptr, hbuf, n, K);
    zero_f(dg, n);
    degree_kernel<<<cdiv(nE,256),256,0,stream>>>(d, w, dg, nE);
    deg_to_dinv<<<cdiv(n,256),256,0,stream>>>(dg, n);
    zero_f(abuf, (long long)n * 32);
    edge_aggregate<<<cdiv((long long)nE*8,256),256,0,stream>>>(s, d, w, dg, hbuf, abuf, nE, 1);
    gcn_finalize<<<cdiv((long long)n*32,256),256,0,stream>>>(abuf, hbuf, dg, bm, out, n);
  };

  // ================= Level 1: local conv ==========================
  gcn(x, srcp, dstp, nullptr, W_lc1, b_lc1, NN, D, E, h1, agg1, deg, x1);
  batch_rowmax<<<Bsz,256,0,stream>>>(x1, xl1, Npb);

  // ================= Global branch 1 (virtual node) ===============
  gen_vnode_edges<<<cdiv(Ev1,256),256,0,stream>>>(vsrc, vdst, Npb, Bsz);
  build_vnode_x<<<cdiv((long long)nv1*32,256),256,0,stream>>>(x1, xv, Npb, Bsz);
  gcn(xv, vsrc, vdst, nullptr, W_gc1, b_gc1, nv1, 32, Ev1, hv, aggv, degv, xgv);
  extract_last<<<cdiv(Bsz*32,256),256,0,stream>>>(xgv, xg1, Npb, Bsz);

  // ================= SAG pool 1 ===================================
  zero_f(agg1, NNH);
  edge_aggregate<<<cdiv((long long)E*8,256),256,0,stream>>>(srcp, dstp, nullptr, deg, x1, agg1, E, 0);
  score_kernel<<<cdiv(NN,256),256,0,stream>>>(agg1, x1, Wrel1, brel1, Wroot1, score, NN);
  copy_f<<<cdiv(NN,256),256,0,stream>>>(score, skey, NN);
  iota_i<<<cdiv(NN,256),256,0,stream>>>(sidx, NN);
  bitonic(skey, sidx, NN);
  pool_gather<<<cdiv((long long)k1*32,256),256,0,stream>>>(x1, skey, sidx, x2, k1);
  fill_i<<<cdiv(NN,256),256,0,stream>>>(inv, -1, NN);
  scatter_inv<<<cdiv(k1,256),256,0,stream>>>(sidx, inv, k1);
  edge_remap<<<cdiv(E,256),256,0,stream>>>(srcp, dstp, nullptr, inv, s2, d2, w2, E);

  // ================= Level 2: local conv ==========================
  gcn(x2, s2, d2, w2, W_lc2, b_lc2, k1, 32, E, h2, agg2, deg, x2p);
  batch_rowmax<<<Bsz,256,0,stream>>>(x2p, xl2, k1 / Bsz);

  // ================= Global branch 2 ==============================
  const int np2 = k1 / Bsz, nv2 = (np2 + 1) * Bsz, Ev2 = np2 * Bsz;
  gen_vnode_edges<<<cdiv(Ev2,256),256,0,stream>>>(vsrc, vdst, np2, Bsz);
  build_vnode_x<<<cdiv((long long)nv2*32,256),256,0,stream>>>(x2p, xv, np2, Bsz);
  gcn(xv, vsrc, vdst, nullptr, W_gc2, b_gc2, nv2, 32, Ev2, hv, aggv, degv, xgv);
  extract_last<<<cdiv(Bsz*32,256),256,0,stream>>>(xgv, xg2, np2, Bsz);

  // ================= SAG pool 2 (only x3 used downstream) =========
  zero_f(agg2, (long long)k1 * 32);
  edge_aggregate<<<cdiv((long long)E*8,256),256,0,stream>>>(s2, d2, w2, deg, x2p, agg2, E, 0);
  score_kernel<<<cdiv(k1,256),256,0,stream>>>(agg2, x2p, Wrel2, brel2, Wroot2, score, k1);
  copy_f<<<cdiv(k1,256),256,0,stream>>>(score, skey, k1);
  iota_i<<<cdiv(k1,256),256,0,stream>>>(sidx, k1);
  bitonic(skey, sidx, k1);
  pool_gather<<<cdiv((long long)k2n*32,256),256,0,stream>>>(x2p, skey, sidx, x3, k2n);
  batch_rowmean<<<Bsz,256,0,stream>>>(x3, xm, k2n / Bsz);

  // ================= Heads ========================================
  small_linear<<<1,256,0,stream>>>(xm, W_l1, b_l1, t8, 16, 32, 8, 1);
  small_linear<<<1,256,0,stream>>>(t8, W_l2, b_l2, obuf, 16, 8, 2, 0);
  concat2<<<cdiv(16*64,256),256,0,stream>>>(xl1, xl2, cat64a, 16, 32, 32);
  small_linear<<<1,256,0,stream>>>(cat64a, W_ll1, b_ll1, t8, 16, 64, 8, 1);
  small_linear<<<1,256,0,stream>>>(t8, W_ll2, b_ll2, xlbuf, 16, 8, 2, 0);
  concat2<<<cdiv(16*64,256),256,0,stream>>>(xg1, xg2, cat64b, 16, 32, 32);
  small_linear<<<1,256,0,stream>>>(cat64b, W_gl1, b_gl1, t8, 16, 64, 8, 1);
  small_linear<<<1,256,0,stream>>>(t8, W_gl2, b_gl2, xgbuf, 16, 8, 2, 0);
  pack_out<<<1,128,0,stream>>>(obuf, xlbuf, xgbuf, (float*)d_out);
}
